// ResidualMambaEncoder_29910152249597
// MI455X (gfx1250) — matrix operand order repaired and checked by reading, not compiled, not run on hardware
//
#include <hip/hip_runtime.h>
#include <hip/hip_bf16.h>
#include <math.h>

// =====================================================================
// ResidualMambaEncoder for MI455X (gfx1250, wave32, WMMA).
// v3: all WMMA operands staged in f16 (converted once at the producer,
// not in the GEMM inner loop).  Main K loop is pure global_load_b128
// (f16) -> v_wmma_f32_16x16x32_f16, no cvt, no branches.  64x16 output
// per wave (4 accumulators reuse one B fragment).  Weights converted to
// f16 once per launch into a pool in d_ws.  Workspace ~72 MB.
// =====================================================================

typedef __attribute__((ext_vector_type(16))) _Float16 v16h;
typedef __attribute__((ext_vector_type(8)))  _Float16 v8h;
typedef __attribute__((ext_vector_type(8)))  float    v8f;

__device__ __forceinline__ float leakyf(float v)    { return v >= 0.f ? v : 0.01f * v; }
__device__ __forceinline__ float siluf(float v)     { return v / (1.f + __expf(-v)); }
__device__ __forceinline__ float softplusf(float v) { return fmaxf(v, 0.f) + log1pf(__expf(-fabsf(v))); }

__device__ __forceinline__ v16h cat8(v8h lo, v8h hi)
{
    return __builtin_shufflevector(lo, hi, 0, 1, 2, 3, 4, 5, 6, 7,
                                   8, 9, 10, 11, 12, 13, 14, 15);
}

// ----------------------- f32 -> f16 convert --------------------------
__global__ void f32_to_f16_kernel(const float* __restrict__ in,
                                  _Float16* __restrict__ out, int n)
{
    int i = blockIdx.x * 256 + threadIdx.x;
    if (i < n) out[i] = (_Float16)in[i];
}

// ---------------------------------------------------------------------
// WMMA GEMM: C[M,N] = act(A[M,K](lda) @ W[N,K]^T(ldw) + bias), A/W f16.
// Optional f16 mirror store C16.  One wave computes (16*MT) x 16.
// Main loop: unconditional v8h (b128) loads from row/col-clamped
// addresses; OOB rows/cols only feed unstored outputs.  Requires
// lda%8==0 && ldw%8==0 whenever K>=32 (all call sites comply); K tail
// uses branchless clamp+select scalar f16 loads.
// Fragment layouts per ISA 7.12.2 (16-bit A 16x32, B 32x16).
// ---------------------------------------------------------------------
template<int MT>
__global__ __launch_bounds__(32)
void gemm_wmma_kernel(const _Float16* __restrict__ A, int lda,
                      const _Float16* __restrict__ W, int ldw,
                      const float* __restrict__ bias,
                      float* __restrict__ C, _Float16* __restrict__ C16,
                      int ldc, int M, int N, int K, int act, int outT)
{
    const int  lane = threadIdx.x;
    const bool hiH  = lane >= 16;
    const int  lm   = lane & 15;
    const int  m_base = blockIdx.x * (16 * MT);
    const int  n0     = blockIdx.y * 16;
    const int  nB     = n0 + lm;
    const int  nBc    = min(nB, N - 1);

    const _Float16* arow[MT];
#pragma unroll
    for (int t = 0; t < MT; ++t)
        arow[t] = A + (size_t)min(m_base + 16 * t + lm, M - 1) * lda;
    const _Float16* wrow = W + (size_t)nBc * ldw;

    const int lowB  = hiH ? 8  : 0;    // A halves 0..7  : K=lowB..lowB+7
    const int highB = hiH ? 24 : 16;   // A halves 8..15 : K=highB..highB+7
    const int bOff  = hiH ? 16 : 0;    // B halves: same K as A (lowB/highB)

    v8f acc[MT] = {};

    const int K32 = K & ~31;
    for (int k0 = 0; k0 < K32; k0 += 32) {
        v16h b = cat8(*(const v8h*)(wrow + k0 + lowB),
                      *(const v8h*)(wrow + k0 + highB));
#pragma unroll
        for (int t = 0; t < MT; ++t) {
            v16h a = cat8(*(const v8h*)(arow[t] + k0 + lowB),
                          *(const v8h*)(arow[t] + k0 + highB));
            acc[t] = __builtin_amdgcn_wmma_f32_16x16x32_f16(
                         false, a, false, b, (short)0, acc[t], false, false);
        }
    }

    if (K32 < K) {                       // scalar branchless tail
        v16h b;
#pragma unroll
        for (int i = 0; i < 16; ++i) {
            int k = (i < 8 ? lowB + i : highB + (i - 8)) + K32;
            _Float16 v = wrow[min(k, K - 1)];
            b[i] = (k < K) ? v : (_Float16)0.f;
        }
#pragma unroll
        for (int t = 0; t < MT; ++t) {
            v16h a;
#pragma unroll
            for (int i = 0; i < 16; ++i) {
                int k = (i < 8 ? lowB + i : highB + (i - 8)) + K32;
                _Float16 v = arow[t][min(k, K - 1)];
                a[i] = (k < K) ? v : (_Float16)0.f;
            }
            acc[t] = __builtin_amdgcn_wmma_f32_16x16x32_f16(
                         false, a, false, b, (short)0, acc[t], false, false);
        }
    }

#pragma unroll
    for (int t = 0; t < MT; ++t) {
#pragma unroll
        for (int i = 0; i < 8; ++i) {
            int m = m_base + 16 * t + i + (hiH ? 8 : 0);
            if (m < M && nB < N) {
                float v = acc[t][i];
                if (bias)     v += bias[nB];
                if (act == 1) v = softplusf(v);
                size_t o = outT ? (size_t)nB * ldc + m : (size_t)m * ldc + nB;
                C[o] = v;
                if (C16) C16[o] = (_Float16)v;
            }
        }
    }
}

// ---------------------------------------------------------------------
// Implicit-GEMM conv2d on WMMA.  X f16 [Ci,Hi,Wi], W f16 row-major
// [Co, Ci*KH*KW], Y f32 [Co,Ho,Wo]+bias.  1x1 main loop unconditional;
// 3x3 pad checks are cndmask-selects on clamped addresses.  W vector
// loads need K%8==0 when K>=32 (holds for Ci*9, Ci%32==0 and K=Ci).
// ---------------------------------------------------------------------
template<int KH, int KW, int MT>
__global__ __launch_bounds__(32)
void conv2d_wmma_kernel(const _Float16* __restrict__ X,
                        const _Float16* __restrict__ W,
                        const float* __restrict__ bias,
                        float* __restrict__ Y,
                        int Ci, int Hi, int Wi,
                        int Co, int Ho, int Wo,
                        int stride, int pad)
{
    const int  KHW = KH * KW;
    const int  M = Ho * Wo, N = Co, K = Ci * KHW;
    const int  lane = threadIdx.x;
    const bool hiH  = lane >= 16;
    const int  lm   = lane & 15;
    const int  m_base = blockIdx.x * (16 * MT);
    const int  n0     = blockIdx.y * 16;
    const int  nB     = n0 + lm;
    const int  nBc    = min(nB, N - 1);

    int oh[MT], ow[MT];
#pragma unroll
    for (int t = 0; t < MT; ++t) {
        int mA = min(m_base + 16 * t + lm, M - 1);
        oh[t] = mA / Wo;
        ow[t] = mA - oh[t] * Wo;
    }

    const int lowB  = hiH ? 8  : 0;
    const int highB = hiH ? 24 : 16;
    const int bOff  = hiH ? 16 : 0;
    const _Float16* wrow = W + (size_t)nBc * K;

    v8f acc[MT] = {};

    const int K32 = K & ~31;
    for (int k0 = 0; k0 < K32; k0 += 32) {
        v16h b = cat8(*(const v8h*)(wrow + k0 + lowB),
                      *(const v8h*)(wrow + k0 + highB));
#pragma unroll
        for (int t = 0; t < MT; ++t) {
            v16h a;
#pragma unroll
            for (int i = 0; i < 16; ++i) {
                int k = (i < 8 ? lowB + i : highB + (i - 8)) + k0;
                if (KH == 1 && KW == 1) {
                    // pad==0; clamped rows keep oh*stride < Hi.
                    a[i] = X[((size_t)k * Hi + oh[t] * stride) * Wi
                             + ow[t] * stride];
                } else {
                    int ci  = k / KHW;
                    int rem = k - ci * KHW;
                    int kh  = rem / KW;
                    int kw  = rem - kh * KW;
                    int ih = oh[t] * stride - pad + kh;
                    int iw = ow[t] * stride - pad + kw;
                    bool ok = (ih >= 0) & (ih < Hi) & (iw >= 0) & (iw < Wi);
                    size_t idx = ok ? ((size_t)ci * Hi + ih) * Wi + iw : 0;
                    _Float16 x = X[idx];
                    a[i] = ok ? x : (_Float16)0.f;
                }
            }
            acc[t] = __builtin_amdgcn_wmma_f32_16x16x32_f16(
                         false, a, false, b, (short)0, acc[t], false, false);
        }
    }

    if (K32 < K) {                       // tail (e.g. stem conv, K=9)
        v16h b;
#pragma unroll
        for (int i = 0; i < 16; ++i) {
            int k = (i < 8 ? lowB + i : highB + (i - 8)) + K32;
            _Float16 v = wrow[min(k, K - 1)];
            b[i] = (k < K) ? v : (_Float16)0.f;
        }
#pragma unroll
        for (int t = 0; t < MT; ++t) {
            v16h a;
#pragma unroll
            for (int i = 0; i < 16; ++i) {
                int k = (i < 8 ? lowB + i : highB + (i - 8)) + K32;
                int kc  = min(k, K - 1);
                int ci  = kc / KHW;
                int rem = kc - ci * KHW;
                int kh  = rem / KW;
                int kw  = rem - kh * KW;
                int ih = oh[t] * stride - pad + kh;
                int iw = ow[t] * stride - pad + kw;
                bool ok = (k < K) & (ih >= 0) & (ih < Hi) & (iw >= 0) & (iw < Wi);
                size_t idx = ok ? ((size_t)ci * Hi + ih) * Wi + iw : 0;
                _Float16 x = X[idx];
                a[i] = ok ? x : (_Float16)0.f;
            }
            acc[t] = __builtin_amdgcn_wmma_f32_16x16x32_f16(
                         false, a, false, b, (short)0, acc[t], false, false);
        }
    }

#pragma unroll
    for (int t = 0; t < MT; ++t) {
#pragma unroll
        for (int i = 0; i < 8; ++i) {
            int m = m_base + 16 * t + i + (hiH ? 8 : 0);
            if (m < M && nB < N)
                Y[(size_t)nB * M + m] = acc[t][i] + bias[nB];   // NCHW
        }
    }
}

// ------------------- InstanceNorm: stats then apply -------------------
__global__ void inorm_stats_kernel(const float* __restrict__ X,
                                   float* __restrict__ stats, int HW)
{
    int c = blockIdx.x, tid = threadIdx.x;
    const float* p = X + (size_t)c * HW;
    float s = 0.f, ss = 0.f;
    for (int i = tid; i < HW; i += 256) { float v = p[i]; s += v; ss += v * v; }
    __shared__ float sh0[256], sh1[256];
    sh0[tid] = s; sh1[tid] = ss; __syncthreads();
    for (int o = 128; o > 0; o >>= 1) {
        if (tid < o) { sh0[tid] += sh0[tid + o]; sh1[tid] += sh1[tid + o]; }
        __syncthreads();
    }
    if (tid == 0) {
        float m = sh0[0] / HW;
        float var = sh1[0] / HW - m * m;
        stats[2 * c]     = m;
        stats[2 * c + 1] = rsqrtf(var + 1e-5f);
    }
}

// y = leaky(inorm(x)) -> f16 (feeds conv2's A operand only)
__global__ void inorm_apply_kernel(const float* __restrict__ X,
                                   const float* __restrict__ stats,
                                   const float* __restrict__ g,
                                   const float* __restrict__ b,
                                   _Float16* __restrict__ Y,
                                   int HW, int total)
{
    int i = blockIdx.x * 256 + threadIdx.x;
    if (i >= total) return;
    int c = i / HW;
    float v = (X[i] - stats[2 * c]) * stats[2 * c + 1] * g[c] + b[c];
    Y[i] = (_Float16)leakyf(v);
}

// out = leaky(inorm(y2)*g+b + idt), stored f32 + f16 mirror
__global__ void res_tail_kernel(const float* __restrict__ Y2,
                                const float* __restrict__ stats,
                                const float* __restrict__ g,
                                const float* __restrict__ b,
                                const float* __restrict__ idt,
                                float* __restrict__ out,
                                _Float16* __restrict__ out16,
                                int HW, int total)
{
    int i = blockIdx.x * 256 + threadIdx.x;
    if (i >= total) return;
    int c = i / HW;
    float v = (Y2[i] - stats[2 * c]) * stats[2 * c + 1] * g[c] + b[c] + idt[i];
    v = leakyf(v);
    out[i]   = v;
    out16[i] = (_Float16)v;
}

// --------- NCHW -> [L,d] transpose fused with LayerNorm(d), f16 out ---
__global__ __launch_bounds__(32)
void ln_transpose_kernel(const float* __restrict__ X,
                         const float* __restrict__ g,
                         const float* __restrict__ b,
                         _Float16* __restrict__ Y, int d, int HW)
{
    int t = blockIdx.x, lane = threadIdx.x;
    float s = 0.f, ss = 0.f;
    for (int c = lane; c < d; c += 32) {
        float v = X[(size_t)c * HW + t];
        s += v; ss += v * v;
    }
#pragma unroll
    for (int o = 16; o > 0; o >>= 1) {
        s  += __shfl_xor(s,  o, 32);
        ss += __shfl_xor(ss, o, 32);
    }
    float mean = s / d;
    float var  = ss / d - mean * mean;
    float rstd = rsqrtf(var + 1e-5f);
    for (int c = lane; c < d; c += 32) {
        float v = X[(size_t)c * HW + t];
        Y[(size_t)t * d + c] = (_Float16)((v - mean) * rstd * g[c] + b[c]);
    }
}

// ----- causal depthwise conv1d (K=4, left pad 3) + SiLU, from xz -----
__global__ void dwconv_silu_kernel(const float* __restrict__ xz,   // [L, 2di]
                                   const float* __restrict__ cw,   // [di,1,4]
                                   const float* __restrict__ cb,
                                   float* __restrict__ xs,         // [L, di] f32
                                   _Float16* __restrict__ xs16,    // f16 mirror
                                   int L, int di)
{
    int i = blockIdx.x * 256 + threadIdx.x;
    if (i >= L * di) return;
    int t = i / di, c = i - t * di;
    float acc = cb[c];
#pragma unroll
    for (int k = 0; k < 4; ++k) {
        int tt = t - 3 + k;
        if (tt >= 0) acc += cw[c * 4 + k] * xz[(size_t)tt * 2 * di + c];
    }
    float v = siluf(acc);
    xs[i]   = v;
    xs16[i] = (_Float16)v;
}

// ---------------------------------------------------------------------
// Selective scan: one wave per channel d, lane n = SSM state (N=32).
// Fuses D-skip and silu(z) gate; writes f16 (only out_proj consumes).
// ---------------------------------------------------------------------
__global__ __launch_bounds__(32)
void mamba_scan_kernel(const float* __restrict__ dt,   // [L,di]
                       const float* __restrict__ xs,   // [L,di]
                       const float* __restrict__ dbl,  // [L,ldd]; B@+r, C@+r+32
                       int r, int ldd,
                       const float* __restrict__ Alog, // [di,32]
                       const float* __restrict__ Dp,   // [di]
                       const float* __restrict__ xz,   // [L,2di] (z at +di)
                       _Float16* __restrict__ Yg,      // [L,di] f16
                       int L, int di)
{
    int d = blockIdx.x, n = threadIdx.x;
    float A  = -__expf(Alog[(size_t)d * 32 + n]);
    float Dv = Dp[d];
    float h  = 0.f;
    for (int t = 0; t < L; ++t) {
        float dtv = dt[(size_t)t * di + d];
        float xv  = xs[(size_t)t * di + d];
        float Bv  = dbl[(size_t)t * ldd + r + n];
        float Cv  = dbl[(size_t)t * ldd + r + 32 + n];
        h = __expf(dtv * A) * h + (dtv * xv) * Bv;
        float contrib = h * Cv;
#pragma unroll
        for (int o = 16; o > 0; o >>= 1)
            contrib += __shfl_down(contrib, o, 32);
        if (n == 0) {
            float y  = contrib + Dv * xv;
            float zv = xz[(size_t)t * 2 * di + di + d];
            Yg[(size_t)t * di + d] = (_Float16)(y * siluf(zv));
        }
    }
}

// ===================== host-side orchestration =======================
struct ResP {
    const float *w1,*b1,*g1,*bb1,*w2,*b2,*g2,*bb2,*w3,*b3;
    const _Float16 *w1h,*w2h,*w3h;
};
struct MamP {
    const float *ln_g,*ln_b,*in_w,*cw,*cb,*xw,*dtw,*dtb,*Alog,*D,*ow;
    const _Float16 *in_wh,*xwh,*dtwh,*owh;
};

static const int GEMM_MT = 4;

static void launch_gemm(hipStream_t s, const _Float16* A, int lda,
                        const _Float16* W, int ldw, const float* bias,
                        float* C, _Float16* C16, int ldc,
                        int M, int N, int K, int act, int outT)
{
    dim3 grid((M + 16 * GEMM_MT - 1) / (16 * GEMM_MT), (N + 15) / 16);
    gemm_wmma_kernel<GEMM_MT><<<grid, 32, 0, s>>>(A, lda, W, ldw, bias,
                                                  C, C16, ldc, M, N, K,
                                                  act, outT);
}

static const _Float16* cvt_weight(hipStream_t s, const float* w, int n,
                                  _Float16*& pool)
{
    _Float16* dst = pool;
    pool += (size_t)((n + 7) & ~7);
    f32_to_f16_kernel<<<(n + 255) / 256, 256, 0, s>>>(w, dst, n);
    return dst;
}

static void run_resblock(hipStream_t s, const ResP& p,
                         const float* in, const _Float16* in16,
                         int Ci, int Hi, int Wi, int Co, int stride,
                         float* out, _Float16* out16,
                         float* T1, _Float16* T1H, float* T2, float* T3,
                         float* stats)
{
    int Ho = (Hi + 2 - 3) / stride + 1;
    int Wo = (Wi + 2 - 3) / stride + 1;
    int HW = Ho * Wo;
    dim3 g((HW + 63) / 64, (Co + 15) / 16);
    int tot = Co * HW;

    conv2d_wmma_kernel<3, 3, 4><<<g, 32, 0, s>>>(in16, p.w1h, p.b1, T1,
                                                 Ci, Hi, Wi, Co, Ho, Wo, stride, 1);
    inorm_stats_kernel<<<Co, 256, 0, s>>>(T1, stats, HW);
    inorm_apply_kernel<<<(tot + 255) / 256, 256, 0, s>>>(T1, stats, p.g1, p.bb1,
                                                         T1H, HW, tot);
    conv2d_wmma_kernel<3, 3, 4><<<g, 32, 0, s>>>(T1H, p.w2h, p.b2, T2,
                                                 Co, Ho, Wo, Co, Ho, Wo, 1, 1);
    inorm_stats_kernel<<<Co, 256, 0, s>>>(T2, stats, HW);
    const float* idt = in;
    if (p.w3) {
        conv2d_wmma_kernel<1, 1, 4><<<g, 32, 0, s>>>(in16, p.w3h, p.b3, T3,
                                                     Ci, Hi, Wi, Co, Ho, Wo, stride, 0);
        idt = T3;
    }
    res_tail_kernel<<<(tot + 255) / 256, 256, 0, s>>>(T2, stats, p.g2, p.bb2,
                                                      idt, out, out16, HW, tot);
}

static void run_mamba(hipStream_t s, const MamP& p, const float* in,
                      float* out, _Float16* out16,
                      int d, int H, int W,
                      _Float16* M_LN16, float* M_XZ,
                      float* M_XS, _Float16* M_XS16,
                      float* M_DBL, _Float16* M_DBL16,
                      float* M_DT, _Float16* M_Y16)
{
    int L = H * W, di = 2 * d, r = (d + 15) / 16, ldd = r + 64;

    ln_transpose_kernel<<<L, 32, 0, s>>>(in, p.ln_g, p.ln_b, M_LN16, d, L);
    // xz = ln(x) @ in_proj_w^T                 [L, 2di]
    launch_gemm(s, M_LN16, d, p.in_wh, d, nullptr,
                M_XZ, nullptr, 2 * di, L, 2 * di, d, 0, 0);
    // xs = silu(depthwise causal conv(xa))     [L, di]
    int tot = L * di;
    dwconv_silu_kernel<<<(tot + 255) / 256, 256, 0, s>>>(M_XZ, p.cw, p.cb,
                                                         M_XS, M_XS16, L, di);
    // dbl = xs @ x_proj_w^T                    [L, r+64], + f16 mirror
    launch_gemm(s, M_XS16, di, p.xwh, di, nullptr,
                M_DBL, M_DBL16, ldd, L, ldd, di, 0, 0);
    // dt = softplus(dbl[:, :r] @ dt_proj_w^T + dt_b)  [L, di]  (K=r: tail path)
    launch_gemm(s, M_DBL16, ldd, p.dtwh, r, p.dtb,
                M_DT, nullptr, di, L, di, r, 1, 0);
    // selective scan + gate                    [L, di] f16
    mamba_scan_kernel<<<di, 32, 0, s>>>(M_DT, M_XS, M_DBL, r, ldd,
                                        p.Alog, p.D, M_XZ, M_Y16, L, di);
    // out = gated @ out_proj_w^T, NCHW [d, L] f32 (+f16 mirror for next conv)
    launch_gemm(s, M_Y16, di, p.owh, di, nullptr,
                out, out16, L, L, d, di, 0, 1);
}

extern "C" void kernel_launch(void* const* d_in, const int* in_sizes, int n_in,
                              void* d_out, int out_size, void* d_ws, size_t ws_size,
                              hipStream_t stream)
{
    (void)in_sizes; (void)n_in; (void)out_size; (void)ws_size;

    int idx = 0;
    auto nxt = [&]() -> const float* { return (const float*)d_in[idx++]; };
    const float* x_in = nxt();

    auto readRes = [&](bool proj) {
        ResP p{};
        p.w1 = nxt(); p.b1 = nxt(); p.g1 = nxt(); p.bb1 = nxt();
        p.w2 = nxt(); p.b2 = nxt(); p.g2 = nxt(); p.bb2 = nxt();
        if (proj) { p.w3 = nxt(); p.b3 = nxt(); }
        return p;
    };
    auto readMam = [&]() {
        MamP p{};
        p.ln_g = nxt(); p.ln_b = nxt(); p.in_w = nxt(); p.cw = nxt();
        p.cb = nxt(); p.xw = nxt(); p.dtw = nxt(); p.dtb = nxt();
        p.Alog = nxt(); p.D = nxt(); p.ow = nxt();
        return p;
    };

    ResP stem  = readRes(true);
    ResP st[4] = { readRes(false), readRes(true), readRes(true), readRes(true) };
    MamP mm[4] = { readMam(), readMam(), readMam(), readMam() };

    // -------- workspace layout: f32 region then f16 region --------
    float* ws = (float*)d_ws;
    const size_t AMAX = 819200;               // 32*160*160 (largest activation)
    float* ACT_A = ws;                        // mamba outputs / stage input
    float* ACT_B = ws + AMAX;                 // resblock outputs
    float* T1    = ws + 2 * AMAX;
    float* T2    = ws + 3 * AMAX;
    float* T3    = ws + 4 * AMAX;
    float* STATS = ws + 5 * AMAX;             // 1024
    float* M_XZ  = STATS + 1024;              // L*2di     <= 3276800
    float* M_XS  = M_XZ  + 3276800;           // L*di      <= 1638400
    float* M_DBL = M_XS  + 1638400;           // L*(r+64)  <= 1689600
    float* M_DT  = M_DBL + 2048000;           // L*di      <= 1638400
    float* FEND  = M_DT  + 1638400;           // = ws + 12,698,624 floats

    _Float16* h = (_Float16*)FEND;            // 16B-aligned (offset%4==0 floats)
    _Float16* ACT_A16 = h;            h += AMAX;
    _Float16* ACT_B16 = h;            h += AMAX;     // written, unused (cheap)
    _Float16* T1H     = h;            h += AMAX;
    _Float16* X16     = h;            h += 25600;
    _Float16* M_LN16  = h;            h += AMAX;
    _Float16* M_XS16  = h;            h += 1638400;
    _Float16* M_DBL16 = h;            h += 2048000;
    _Float16* M_Y16   = h;            h += 1638400;
    _Float16* wpool   = h;                           // ~1.84M halves of weights

    // ---- one-time-per-launch f16 conversions (deterministic) ----
    f32_to_f16_kernel<<<(25600 + 255) / 256, 256, 0, stream>>>(x_in, X16, 25600);

    const int dims[4] = { 32, 64, 128, 256 };
    const int cin[4]  = { 32, 32, 64, 128 };

    stem.w1h = cvt_weight(stream, stem.w1, 32 * 1 * 9, wpool);
    stem.w2h = cvt_weight(stream, stem.w2, 32 * 32 * 9, wpool);
    stem.w3h = cvt_weight(stream, stem.w3, 32 * 1, wpool);
    for (int s = 0; s < 4; ++s) {
        int Ci = cin[s], Co = dims[s];
        st[s].w1h = cvt_weight(stream, st[s].w1, Co * Ci * 9, wpool);
        st[s].w2h = cvt_weight(stream, st[s].w2, Co * Co * 9, wpool);
        if (st[s].w3)
            st[s].w3h = cvt_weight(stream, st[s].w3, Co * Ci, wpool);
    }
    for (int s = 0; s < 4; ++s) {
        int d = dims[s], di = 2 * d, r = (d + 15) / 16;
        mm[s].in_wh = cvt_weight(stream, mm[s].in_w, 2 * di * d, wpool);
        mm[s].xwh   = cvt_weight(stream, mm[s].xw, (r + 64) * di, wpool);
        mm[s].dtwh  = cvt_weight(stream, mm[s].dtw, di * r, wpool);
        mm[s].owh   = cvt_weight(stream, mm[s].ow, d * di, wpool);
    }

    // ---------------------- network ----------------------
    // stem: 1 -> 32, stride 1, 160x160
    run_resblock(stream, stem, x_in, X16, 1, 160, 160, 32, 1,
                 ACT_A, ACT_A16, T1, T1H, T2, T3, STATS);

    const int strides[4] = { 1, 2, 2, 2 };
    const int hs[5]      = { 160, 160, 80, 40, 20 };

    for (int s = 0; s < 4; ++s) {
        int Hi = hs[s], Ho = hs[s + 1];
        run_resblock(stream, st[s], ACT_A, ACT_A16, cin[s], Hi, Hi,
                     dims[s], strides[s], ACT_B, ACT_B16,
                     T1, T1H, T2, T3, STATS);
        float*    mout   = (s == 3) ? (float*)d_out : ACT_A;
        _Float16* mout16 = (s == 3) ? nullptr       : ACT_A16;
        run_mamba(stream, mm[s], ACT_B, mout, mout16, dims[s], Ho, Ho,
                  M_LN16, M_XZ, M_XS, M_XS16, M_DBL, M_DBL16, M_DT, M_Y16);
    }
}
